// NeuromorphicAttention_5076651344509
// MI455X (gfx1250) — matrix operand order repaired and checked by reading, not compiled, not run on hardware
//
#include <hip/hip_runtime.h>
#include <hip/hip_bf16.h>
#include <math.h>

typedef __attribute__((ext_vector_type(16))) __bf16 bf16x16;
typedef __attribute__((ext_vector_type(8)))  __bf16 bf16x8;
typedef __attribute__((ext_vector_type(8)))  float  v8f;

#define WMMA_BF16(a, b, c) \
  __builtin_amdgcn_wmma_f32_16x16x32_bf16(false, (a), false, (b), (short)0, (c), false, false)

static __device__ inline bf16x16 join8(bf16x8 lo, bf16x8 hi) {
  bf16x16 r;
#pragma unroll
  for (int e = 0; e < 8; e++) { r[e] = lo[e]; r[8 + e] = hi[e]; }
  return r;
}

// ---------------------------------------------------------------------------
// f32 -> bf16 convert
// ---------------------------------------------------------------------------
__global__ void f32_to_bf16(const float* __restrict__ in, __bf16* __restrict__ out, int n) {
  int i = blockIdx.x * blockDim.x + threadIdx.x;
  if (i < n) out[i] = (__bf16)in[i];
}

// ---------------------------------------------------------------------------
// Fold per-head input weights into projection: Weff[h*64+j, d] = sum_i Win[j,i] * W[h*64+i, d]
// ---------------------------------------------------------------------------
__global__ void fold_w(const float* __restrict__ Win, const float* __restrict__ W,
                       __bf16* __restrict__ Weff) {
  int idx = blockIdx.x * blockDim.x + threadIdx.x;   // 0 .. 1M-1
  int c = idx >> 10;
  int d = idx & 1023;
  int h = c >> 6, j = c & 63;
  const float* wi = Win + j * 64;
  const float* wr = W + (size_t)(h * 64) * 1024 + d;
  float s = 0.f;
#pragma unroll 8
  for (int i = 0; i < 64; i++) s += wi[i] * wr[(size_t)i * 1024];
  Weff[idx] = (__bf16)s;
}

// ---------------------------------------------------------------------------
// WMMA GEMM: out = A[M,K] @ Bw[N,K]^T  (+ bias in mode 1)
// mode 0: scatter into cur layout [S, 64, 64] (neuron = b*16+h)
// mode 1: plain [M,N] fp32 + bias
// block = 256 threads (8 waves), tile 128(M) x 64(N)
// ---------------------------------------------------------------------------
__global__ __launch_bounds__(256)
void gemm_bf16_nt(const __bf16* __restrict__ A, const __bf16* __restrict__ Bw,
                  const float* __restrict__ bias, float* __restrict__ out,
                  int M, int N, int K, int mode) {
  const int tid  = threadIdx.x;
  const int w    = tid >> 5;
  const int lane = tid & 31;
  const int half = (lane >> 4) & 1;
  const int l16  = lane & 15;
  const int mbase = blockIdx.x * 128 + w * 16;
  const int nbase = blockIdx.y * 64;
  const int arow  = mbase + l16;

  v8f acc[4];
#pragma unroll
  for (int n = 0; n < 4; n++)
#pragma unroll
    for (int r = 0; r < 8; r++) acc[n][r] = 0.f;

  for (int kk = 0; kk < K; kk += 32) {
    const __bf16* ap = A + (size_t)arow * K + kk + half * 16;
    bf16x16 afrag = join8(*(const bf16x8*)ap, *(const bf16x8*)(ap + 8));
#pragma unroll
    for (int n = 0; n < 4; n++) {
      int brow = nbase + n * 16 + l16;
      bf16x16 bfrag = *(const bf16x16*)(Bw + (size_t)brow * K + kk + half * 16);
      acc[n] = WMMA_BF16(afrag, bfrag, acc[n]);
    }
  }

#pragma unroll
  for (int n = 0; n < 4; n++) {
    int col = nbase + n * 16 + l16;
#pragma unroll
    for (int r = 0; r < 8; r++) {
      int m = mbase + r + 8 * half;
      float v = acc[n][r];
      if (mode == 0) {
        int b = m >> 10, s = m & 1023;
        int h = col >> 6, j = col & 63;
        out[((size_t)s * 64 + b * 16 + h) * 64 + j] = v;
      } else {
        out[(size_t)m * N + col] = v + bias[col];
      }
    }
  }
}

// ---------------------------------------------------------------------------
// LIF recurrence. 12 blocks (3 lifs x 4 row-blocks of 16 neurons), 128 thr (4 waves).
// Each wave owns a 16x16 (rows x Dh-cols) tile in WMMA C layout; spike history
// double-buffered in LDS as bf16 (spikes are exactly 0/1 -> bf16 exact);
// W_rec B-fragments preloaded in registers (constant across all 1024 steps).
// ---------------------------------------------------------------------------
__global__ __launch_bounds__(128)
void lif_kernel(const float* __restrict__ cur_q, const float* __restrict__ cur_k,
                const float* __restrict__ cur_v,
                const float* __restrict__ Wrq, const float* __restrict__ Wrk,
                const float* __restrict__ Wrv,
                const float* __restrict__ thq, const float* __restrict__ thk,
                const float* __restrict__ thv,
                __bf16* __restrict__ sq, __bf16* __restrict__ sk, __bf16* __restrict__ sv) {
  const int lif = blockIdx.x >> 2;
  const int rb  = blockIdx.x & 3;
  const float* cur  = lif == 0 ? cur_q : lif == 1 ? cur_k : cur_v;
  const float* Wrec = lif == 0 ? Wrq  : lif == 1 ? Wrk  : Wrv;
  const float* th   = lif == 0 ? thq  : lif == 1 ? thk  : thv;
  __bf16* spk       = lif == 0 ? sq   : lif == 1 ? sk   : sv;

  const int tid  = threadIdx.x;
  const int w    = tid >> 5;
  const int lane = tid & 31;
  const int half = (lane >> 4) & 1;
  const int l16  = lane & 15;

  __shared__ __align__(32) __bf16 hist[2][16][64];   // double-buffered spike history
  for (int i = tid; i < 2 * 16 * 64; i += 128) ((__bf16*)hist)[i] = (__bf16)0.f;
  __syncthreads();

  // Preload W_rec^T B-fragments for this wave's 16 output columns (k = 0..63)
  bf16x16 bw[2];
#pragma unroll
  for (int ks = 0; ks < 2; ks++) {
    const float* src = Wrec + (size_t)(w * 16 + l16) * 64 + ks * 32 + half * 16;
#pragma unroll
    for (int e = 0; e < 16; e++) bw[ks][e] = (__bf16)src[e];
  }
  const float theta = th[w * 16 + l16];

  float vmem[8], refr[8], athr[8];
#pragma unroll
  for (int r = 0; r < 8; r++) { vmem[r] = 0.f; refr[r] = 0.f; athr[r] = 1.0f; }

  for (int s = 0; s < 1024; s++) {
    const int cb = s & 1;
    const __bf16* hrow = &hist[cb][l16][0];
    bf16x16 a0 = join8(*(const bf16x8*)(hrow + half * 16),
                       *(const bf16x8*)(hrow + 8 + half * 16));
    bf16x16 a1 = join8(*(const bf16x8*)(hrow + 32 + half * 16),
                       *(const bf16x8*)(hrow + 40 + half * 16));

    v8f acc;
#pragma unroll
    for (int r = 0; r < 8; r++)
      acc[r] = cur[((size_t)s * 64 + rb * 16 + r + 8 * half) * 64 + w * 16 + l16];

    acc = WMMA_BF16(a0, bw[0], acc);   // total = cur + hist @ Wrec^T
    acc = WMMA_BF16(a1, bw[1], acc);

#pragma unroll
    for (int r = 0; r < 8; r++) {
      float v = 0.9f * vmem[r] + acc[r];
      if (refr[r] > 0.f) v = 0.f;                  // refractory masking
      float spike = (v >= athr[r]) ? 1.f : 0.f;
      v = v * (1.f - spike);                       // membrane reset
      vmem[r] = v;
      refr[r] = fmaxf(refr[r] - 1.f, spike * 2.f);
      athr[r] = 0.95f * athr[r] + spike * theta;
      __bf16 sb = (__bf16)spike;
      hist[cb ^ 1][r + 8 * half][w * 16 + l16] = sb;
      spk[((size_t)(rb * 16 + r + 8 * half) * 1024 + s) * 64 + w * 16 + l16] = sb;
    }
    __syncthreads();
  }
}

// ---------------------------------------------------------------------------
// Flash attention over binary spikes. grid (64 = b*16+h, 16 q-blocks of 64),
// 128 threads (4 waves, one 16-row q tile each). Online softmax; P re-layout
// C->A via per-wave LDS; V chunk staged transposed in LDS.
// ---------------------------------------------------------------------------
__global__ __launch_bounds__(128)
void attn_kernel(const __bf16* __restrict__ spk_q, const __bf16* __restrict__ spk_k,
                 const __bf16* __restrict__ spk_v, __bf16* __restrict__ ctx) {
  const int bh  = blockIdx.x;        // b*16 + h
  const int qb  = blockIdx.y * 64;
  const int tid = threadIdx.x;
  const int w    = tid >> 5;
  const int lane = tid & 31;
  const int half = (lane >> 4) & 1;
  const int l16  = lane & 15;
  const int b = bh >> 4, h = bh & 15;

  __shared__ __align__(32) __bf16 Vt[64][64];        // [j][key] transposed V chunk
  __shared__ __align__(32) __bf16 Pbuf[4][16][64];   // per-wave P tile

  const __bf16* Q  = spk_q + (size_t)bh * 1024 * 64;
  const __bf16* Km = spk_k + (size_t)bh * 1024 * 64;
  const __bf16* Vm = spk_v + (size_t)bh * 1024 * 64;

  // Q fragments (A layout), pre-scaled by 1/sqrt(Dh)=0.125 (exact on {0,1})
  bf16x16 qa[2];
  {
    const __bf16* qrow = Q + (size_t)(qb + w * 16 + l16) * 64;
#pragma unroll
    for (int ks = 0; ks < 2; ks++) {
#pragma unroll
      for (int e = 0; e < 16; e++) {
        int k = ks * 32 + half * 16 + e;
        qa[ks][e] = (__bf16)(0.125f * (float)qrow[k]);
      }
    }
  }

  float mrow[8], lrow[8];
  v8f acc[4];
#pragma unroll
  for (int r = 0; r < 8; r++) { mrow[r] = -1e30f; lrow[r] = 0.f; }
#pragma unroll
  for (int n = 0; n < 4; n++)
#pragma unroll
    for (int r = 0; r < 8; r++) acc[n][r] = 0.f;

  for (int kt = 0; kt < 1024; kt += 64) {
    // stage V chunk transposed into LDS
    for (int t = tid; t < 4096; t += 128) {
      int key = t >> 6, j = t & 63;
      Vt[j][key] = Vm[(size_t)(kt + key) * 64 + j];
    }
    __syncthreads();

    // scores S = Qs @ K^T  (16 x 64 per wave); K B-fragments contiguous in global
    v8f sc[4];
#pragma unroll
    for (int n = 0; n < 4; n++) {
#pragma unroll
      for (int r = 0; r < 8; r++) sc[n][r] = 0.f;
      const __bf16* krow = Km + (size_t)(kt + n * 16 + l16) * 64;
      bf16x16 kb0 = *(const bf16x16*)(krow + half * 16);
      bf16x16 kb1 = *(const bf16x16*)(krow + 32 + half * 16);
      sc[n] = WMMA_BF16(qa[0], kb0, sc[n]);
      sc[n] = WMMA_BF16(qa[1], kb1, sc[n]);
    }

    // online softmax (row stats reduced across the 16-lane half-group)
#pragma unroll
    for (int r = 0; r < 8; r++) {
      float vmax = sc[0][r];
#pragma unroll
      for (int n = 1; n < 4; n++) vmax = fmaxf(vmax, sc[n][r]);
#pragma unroll
      for (int msk = 1; msk < 16; msk <<= 1) vmax = fmaxf(vmax, __shfl_xor(vmax, msk));
      float newm  = fmaxf(mrow[r], vmax);
      float scale = __expf(mrow[r] - newm);
      float lsum  = 0.f;
#pragma unroll
      for (int n = 0; n < 4; n++) {
        float p = __expf(sc[n][r] - newm);
        lsum += p;
        Pbuf[w][r + 8 * half][n * 16 + l16] = (__bf16)p;
      }
#pragma unroll
      for (int msk = 1; msk < 16; msk <<= 1) lsum += __shfl_xor(lsum, msk);
      lrow[r] = lrow[r] * scale + lsum;
      mrow[r] = newm;
#pragma unroll
      for (int n = 0; n < 4; n++) acc[n][r] *= scale;
    }

    asm volatile("s_wait_dscnt 0" ::: "memory");   // wave-local Pbuf RAW

    const __bf16* prow = &Pbuf[w][l16][0];
    bf16x16 pa0 = join8(*(const bf16x8*)(prow + half * 16),
                        *(const bf16x8*)(prow + 8 + half * 16));
    bf16x16 pa1 = join8(*(const bf16x8*)(prow + 32 + half * 16),
                        *(const bf16x8*)(prow + 40 + half * 16));

#pragma unroll
    for (int n = 0; n < 4; n++) {
      const __bf16* vtr = &Vt[n * 16 + l16][0];
      bf16x16 vb0 = *(const bf16x16*)(vtr + half * 16);
      bf16x16 vb1 = *(const bf16x16*)(vtr + 32 + half * 16);
      acc[n] = WMMA_BF16(pa0, vb0, acc[n]);
      acc[n] = WMMA_BF16(pa1, vb1, acc[n]);
    }
    __syncthreads();
  }

  // normalize and store ctx (bf16, [B,S,D] row = b*S+s)
#pragma unroll
  for (int r = 0; r < 8; r++) {
    float inv = 1.f / lrow[r];
    int srow = qb + w * 16 + r + 8 * half;
#pragma unroll
    for (int n = 0; n < 4; n++)
      ctx[((size_t)b * 1024 + srow) * 1024 + h * 64 + n * 16 + l16] =
          (__bf16)(acc[n][r] * inv);
  }
}

// ---------------------------------------------------------------------------
extern "C" void kernel_launch(void* const* d_in, const int* in_sizes, int n_in,
                              void* d_out, int out_size, void* d_ws, size_t ws_size,
                              hipStream_t stream) {
  const float* x      = (const float*)d_in[0];
  const float* Wq     = (const float*)d_in[1];
  const float* Wk     = (const float*)d_in[2];
  const float* Wv     = (const float*)d_in[3];
  const float* Wo     = (const float*)d_in[4];
  const float* bo     = (const float*)d_in[5];
  const float* Wq_in  = (const float*)d_in[6];
  const float* Wq_rec = (const float*)d_in[7];
  const float* Wk_in  = (const float*)d_in[8];
  const float* Wk_rec = (const float*)d_in[9];
  const float* Wv_in  = (const float*)d_in[10];
  const float* Wv_rec = (const float*)d_in[11];
  const float* thq    = (const float*)d_in[12];
  const float* thk    = (const float*)d_in[13];
  const float* thv    = (const float*)d_in[14];
  float* out = (float*)d_out;

  char* ws = (char*)d_ws;
  size_t off = 0;
  auto carve = [&](size_t bytes) -> char* {
    char* p = ws + off;
    off += (bytes + 255) & ~(size_t)255;
    return p;
  };

  __bf16* xbf  = (__bf16*)carve(4096ull * 1024 * 2);       // x in bf16
  __bf16* wobf = (__bf16*)carve(1024ull * 1024 * 2);       // Wo in bf16
  __bf16* weq  = (__bf16*)carve(1024ull * 1024 * 2);       // folded proj weights
  __bf16* wek  = (__bf16*)carve(1024ull * 1024 * 2);
  __bf16* wev  = (__bf16*)carve(1024ull * 1024 * 2);
  float*  curq = (float*)carve(1024ull * 64 * 64 * 4);     // [S,64,64] currents
  float*  curk = (float*)carve(1024ull * 64 * 64 * 4);
  float*  curv = (float*)carve(1024ull * 64 * 64 * 4);
  __bf16* sq   = (__bf16*)carve(64ull * 1024 * 64 * 2);    // spikes [BH,S,Dh]
  __bf16* sk   = (__bf16*)carve(64ull * 1024 * 64 * 2);
  __bf16* sv   = (__bf16*)carve(64ull * 1024 * 64 * 2);
  __bf16* ctx  = (__bf16*)carve(4096ull * 1024 * 2);       // [B*S, D] bf16

  f32_to_bf16<<<(4194304 + 255) / 256, 256, 0, stream>>>(x, xbf, 4194304);
  f32_to_bf16<<<(1048576 + 255) / 256, 256, 0, stream>>>(Wo, wobf, 1048576);
  fold_w<<<4096, 256, 0, stream>>>(Wq_in, Wq, weq);
  fold_w<<<4096, 256, 0, stream>>>(Wk_in, Wk, wek);
  fold_w<<<4096, 256, 0, stream>>>(Wv_in, Wv, wev);

  dim3 gg(32, 16);
  gemm_bf16_nt<<<gg, 256, 0, stream>>>(xbf, weq, nullptr, curq, 4096, 1024, 1024, 0);
  gemm_bf16_nt<<<gg, 256, 0, stream>>>(xbf, wek, nullptr, curk, 4096, 1024, 1024, 0);
  gemm_bf16_nt<<<gg, 256, 0, stream>>>(xbf, wev, nullptr, curv, 4096, 1024, 1024, 0);

  lif_kernel<<<12, 128, 0, stream>>>(curq, curk, curv, Wq_rec, Wk_rec, Wv_rec,
                                     thq, thk, thv, sq, sk, sv);

  attn_kernel<<<dim3(64, 16), 128, 0, stream>>>(sq, sk, sv, ctx);

  gemm_bf16_nt<<<gg, 256, 0, stream>>>(ctx, wobf, bo, out, 4096, 1024, 1024, 1);
}
